// NetSimile_3934190044273
// MI455X (gfx1250) — 8-bit matrix operand order repaired and checked by reading, compiled, not run on hardware
//
#include <hip/hip_runtime.h>
#include <hip/hip_bf16.h>

typedef __attribute__((ext_vector_type(16))) int   v16i;
typedef __attribute__((ext_vector_type(8)))  float v8f;

#define NN      4096
#define GG      8
#define MPER    512
#define BLK     128
#define APITCH  144
#define FP8_ONE 0x38

#if defined(__AMDGCN__) && __has_builtin(__builtin_amdgcn_tensor_load_to_lds) && \
    __has_builtin(__builtin_amdgcn_s_wait_tensorcnt)
#define USE_TDM 1
#else
#define USE_TDM 0
#endif

#if USE_TDM
typedef __attribute__((ext_vector_type(4))) unsigned int tdm_v4u;
typedef __attribute__((ext_vector_type(8))) int          tdm_v8i;
typedef __attribute__((ext_vector_type(4))) int          tdm_v4i;

__device__ __forceinline__ void tdm_tile_load(unsigned ldsOff,
                                              const unsigned char* gsrc) {
    unsigned long long ga = (unsigned long long)(size_t)gsrc;
    tdm_v4u g0 = { 1u,
                   ldsOff,
                   (unsigned)ga,
                   (unsigned)((ga >> 32) & 0x01FFFFFFull) | 0x80000000u };
    tdm_v8i g1 = { (int)((1u << 20) | (4u << 22) | (3u << 25)),
                   (int)(512u << 16),
                   (int)(512u << 16),
                   (int)(128u << 16),
                   128,
                   512,
                   0, 0 };
    tdm_v4i z4 = { 0, 0, 0, 0 };
    tdm_v8i z8 = { 0, 0, 0, 0, 0, 0, 0, 0 };
    __builtin_amdgcn_tensor_load_to_lds(g0, g1, z4, z4, z8, 0);
}
#endif

__global__ __launch_bounds__(256) void k1_deg_pack(
    const float* __restrict__ A, unsigned char* __restrict__ m1,
    float* __restrict__ deg, float* __restrict__ f4acc,
    float* __restrict__ m2cnt, float* __restrict__ m2deg)
{
    const int w    = threadIdx.x >> 5;
    const int lane = threadIdx.x & 31;
    const int n    = blockIdx.x * 8 + w;
    const int g    = n >> 9;
    const int r    = n & 511;
    const int lo   = g << 9;

    float s = 0.f;
    unsigned char* mrow = m1 + (size_t)g * MPER * MPER + (size_t)r * MPER;
    const float*   arow = A + (size_t)n * NN + lo;
#pragma unroll
    for (int it = 0; it < 16; ++it) {
        int i = lane + it * 32;
        float a = arow[i];
        s += a;
        mrow[i] = (a != 0.f || i == r) ? (unsigned char)FP8_ONE : (unsigned char)0;
    }
#pragma unroll
    for (int m = 16; m >= 1; m >>= 1) s += __shfl_xor(s, m, 32);
    if (lane == 0) {
        deg[n]   = s;
        f4acc[n] = 0.f; m2cnt[n] = 0.f; m2deg[n] = 0.f;
    }
}

__global__ __launch_bounds__(256) void k2_wmma(
    const unsigned char* __restrict__ m1, const float* __restrict__ deg,
    float* __restrict__ f4acc, float* __restrict__ m2cnt, float* __restrict__ m2deg)
{
    __shared__ __align__(16) unsigned char At[2][BLK * APITCH];
    __shared__ __align__(16) unsigned char Bt[2][BLK * APITCH];
    __shared__ float degs[BLK];

    const int g     = blockIdx.z;
    const int rBase = blockIdx.y * BLK;
    const int cBase = blockIdx.x * BLK;
    const unsigned char* M = m1 + (size_t)g * MPER * MPER;

    const int t    = threadIdx.x;
    const int w    = t >> 5;
    const int lane = t & 31;
    const int h    = lane >> 4;
    const int l4   = lane & 15;

    if (t < BLK) degs[t] = deg[g * MPER + cBase + t];

    v8f acc[8];
#pragma unroll
    for (int i = 0; i < 8; ++i) acc[i] = (v8f){};

#if USE_TDM
    const unsigned atOff0 = (unsigned)(size_t)(void*)&At[0][0];
    const unsigned atOff1 = (unsigned)(size_t)(void*)&At[1][0];
    const unsigned btOff0 = (unsigned)(size_t)(void*)&Bt[0][0];
    const unsigned btOff1 = (unsigned)(size_t)(void*)&Bt[1][0];
    if (w == 0) {
        tdm_tile_load(atOff0, M + (size_t)rBase * MPER);
        tdm_tile_load(btOff0, M + (size_t)cBase * MPER);
    }
#endif

    for (int kt = 0; kt < 4; ++kt) {
        const int cur = kt & 1;
        __syncthreads();
#if USE_TDM
        if (w == 0) {
            if (kt < 3) {
                const int k1o = (kt + 1) * 128;
                tdm_tile_load(cur ? atOff0 : atOff1, M + (size_t)rBase * MPER + k1o);
                tdm_tile_load(cur ? btOff0 : btOff1, M + (size_t)cBase * MPER + k1o);
                __builtin_amdgcn_s_wait_tensorcnt((short)2);
            } else {
                __builtin_amdgcn_s_wait_tensorcnt((short)0);
            }
        }
#else
        {
            const int k0 = kt * 128;
            for (int i = t; i < 1024; i += 256) {
                int row = i >> 3, cc = (i & 7) * 16;
                *(uint4*)&At[cur][row * APITCH + cc] =
                    *(const uint4*)&M[(size_t)(rBase + row) * MPER + k0 + cc];
                *(uint4*)&Bt[cur][row * APITCH + cc] =
                    *(const uint4*)&M[(size_t)(cBase + row) * MPER + k0 + cc];
                if (kt < 3)
                    __builtin_prefetch(&M[(size_t)(rBase + row) * MPER + k0 + 128 + cc], 0, 0);
            }
        }
#endif
        __syncthreads();

        v16i af;
        {
            const unsigned char* ar = &At[cur][(w * 16 + l4) * APITCH];
#pragma unroll
            for (int c = 0; c < 2; ++c)
#pragma unroll
                for (int p = 0; p < 4; ++p) {
                    uint2 u = *(const uint2*)(ar + c * 64 + (p >> 1) * 32 + h * 16 + (p & 1) * 8);
                    af[c * 8 + p * 2]     = (int)u.x;
                    af[c * 8 + p * 2 + 1] = (int)u.y;
                }
        }
#pragma unroll
        for (int tn = 0; tn < 8; ++tn) {
            v16i bf;
            const unsigned char* br = &Bt[cur][(tn * 16 + l4) * APITCH];
#pragma unroll
            for (int jj = 0; jj < 4; ++jj) {
                uint4 q = *(const uint4*)(br + jj * 32 + h * 16);
                bf[jj * 4 + 0] = (int)q.x; bf[jj * 4 + 1] = (int)q.y;
                bf[jj * 4 + 2] = (int)q.z; bf[jj * 4 + 3] = (int)q.w;
            }
            acc[tn] = __builtin_amdgcn_wmma_f32_16x16x128_fp8_fp8(
                af, bf, (short)0, acc[tn], false, false);
        }
    }

    __syncthreads();
#if USE_TDM
    if (w == 0) {
        tdm_tile_load((unsigned)(size_t)(void*)&At[0][0],
                      M + (size_t)rBase * MPER + cBase);
        __builtin_amdgcn_s_wait_tensorcnt((short)0);
    }
#else
    for (int i = t; i < 1024; i += 256) {
        int row = i >> 3, cc = (i & 7) * 16;
        *(uint4*)&At[0][row * APITCH + cc] =
            *(const uint4*)&M[(size_t)(rBase + row) * MPER + cBase + cc];
    }
#endif
    __syncthreads();

#pragma unroll
    for (int j = 0; j < 8; ++j) {
        const int rowL = w * 16 + j + 8 * h;
        float sf4 = 0.f, scnt = 0.f, sdeg = 0.f;
#pragma unroll
        for (int tn = 0; tn < 8; ++tn) {
            float c    = acc[tn][j];
            int   colL = tn * 16 + l4;
            float m2v  = (c > 0.5f) ? 1.f : 0.f;
            scnt += m2v;
            sdeg += m2v * degs[colL];
            if (At[0][rowL * APITCH + colL]) sf4 += c;
        }
#pragma unroll
        for (int m = 1; m <= 8; m <<= 1) {
            sf4  += __shfl_xor(sf4,  m, 32);
            scnt += __shfl_xor(scnt, m, 32);
            sdeg += __shfl_xor(sdeg, m, 32);
        }
        if (l4 == 0) {
            int node = g * MPER + rBase + rowL;
            atomicAdd(&f4acc[node], sf4);
            atomicAdd(&m2cnt[node], scnt);
            atomicAdd(&m2deg[node], sdeg);
        }
    }
}

__global__ __launch_bounds__(256) void k3_feats(
    const float* __restrict__ A, const float* __restrict__ deg,
    const float* __restrict__ f4acc, const float* __restrict__ m2cnt,
    const float* __restrict__ m2deg, float* __restrict__ feat)
{
    const int w = threadIdx.x >> 5, lane = threadIdx.x & 31;
    const int n = blockIdx.x * 8 + w;
    const int lo = (n >> 9) << 9;
    float s = 0.f;
    const float* arow = A + (size_t)n * NN + lo;
#pragma unroll
    for (int it = 0; it < 16; ++it) {
        int i = lane + it * 32;
        s += arow[i] * deg[lo + i];
    }
#pragma unroll
    for (int m = 16; m >= 1; m >>= 1) s += __shfl_xor(s, m, 32);
    if (lane == 0) {
        float d   = deg[n];
        float f2  = (d > 0.f) ? s / d : 0.f;
        float f4r = f4acc[n] - d - 1.f;
        float f1  = (d > 1.f) ? (f4r - d) / (d * (d - 1.f)) * 2.f : 0.f;
        float* fr = feat + (size_t)n * 7;
        fr[0] = d;
        fr[1] = f1;
        fr[2] = f2;
        fr[4] = f4r * 0.5f;
        fr[5] = m2deg[n] - 2.f * f4r;
        fr[6] = m2cnt[n] - d - 1.f;
    }
}

__global__ __launch_bounds__(256) void k4_f3(
    const float* __restrict__ A, const float* __restrict__ deg,
    float* __restrict__ feat)
{
    const int w = threadIdx.x >> 5, lane = threadIdx.x & 31;
    const int n = blockIdx.x * 8 + w;
    const int lo = (n >> 9) << 9;
    float s = 0.f;
    const float* arow = A + (size_t)n * NN + lo;
#pragma unroll
    for (int it = 0; it < 16; ++it) {
        int i = lane + it * 32;
        s += arow[i] * feat[(size_t)(lo + i) * 7 + 1];
    }
#pragma unroll
    for (int m = 16; m >= 1; m >>= 1) s += __shfl_xor(s, m, 32);
    if (lane == 0) {
        float d = deg[n];
        feat[(size_t)n * 7 + 3] = (d > 0.f) ? s / d : 0.f;
    }
}

__global__ __launch_bounds__(256) void k5_stats(
    const float* __restrict__ feat, float* __restrict__ out)
{
    __shared__ float x[MPER];
    __shared__ float red[256];
    __shared__ float med;
    const int gi = blockIdx.x / 7, fi = blockIdx.x % 7;
    const int t  = threadIdx.x;

    x[t]       = feat[(size_t)(gi * MPER + t)       * 7 + fi];
    x[t + 256] = feat[(size_t)(gi * MPER + t + 256) * 7 + fi];
    __syncthreads();

    auto blockSum = [&](float v) -> float {
        __syncthreads();
        red[t] = v; __syncthreads();
        for (int s = 128; s > 0; s >>= 1) {
            if (t < s) red[t] += red[t + s];
            __syncthreads();
        }
        return red[0];
    };

    float mean = blockSum(x[t] + x[t + 256]) * (1.f / MPER);

    float p2 = 0.f, p3 = 0.f, p4 = 0.f;
#pragma unroll
    for (int e = 0; e < 2; ++e) {
        float c = x[t + e * 256] - mean;
        float c2 = c * c;
        p2 += c2; p3 += c2 * c; p4 += c2 * c2;
    }
    float m2 = blockSum(p2) * (1.f / MPER);
    float m3 = blockSum(p3) * (1.f / MPER);
    float m4 = blockSum(p4) * (1.f / MPER);

#pragma unroll
    for (int e = 0; e < 2; ++e) {
        float v = x[t + e * 256];
        int lt = 0, eq = 0;
        for (int k = 0; k < MPER; ++k) {
            float xv = x[k];
            lt += (xv < v);
            eq += (xv == v);
        }
        if (lt <= 255 && 255 < lt + eq) med = v;
    }
    __syncthreads();

    if (t == 0) {
        const float eps = 1e-4f;
        float* og = out + gi * 35;
        og[fi]      = mean;
        og[7 + fi]  = med;
        og[14 + fi] = sqrtf(m2);
        og[21 + fi] = m3 / fmaxf(m2 * sqrtf(m2), eps);
        og[28 + fi] = m4 / fmaxf(m2 * m2, eps);
    }
}

extern "C" void kernel_launch(void* const* d_in, const int* in_sizes, int n_in,
                              void* d_out, int out_size, void* d_ws, size_t ws_size,
                              hipStream_t stream) {
    const float* A = (const float*)d_in[0];
    float* out = (float*)d_out;

    unsigned char* ws = (unsigned char*)d_ws;
    unsigned char* m1    = ws;
    float*         deg   = (float*)(ws + 2097152);
    float*         f4acc = (float*)(ws + 2097152 + 16384);
    float*         m2cnt = (float*)(ws + 2097152 + 32768);
    float*         m2deg = (float*)(ws + 2097152 + 49152);
    float*         feat  = (float*)(ws + 2097152 + 65536);

    k1_deg_pack<<<512, 256, 0, stream>>>(A, m1, deg, f4acc, m2cnt, m2deg);
    k2_wmma<<<dim3(4, 4, 8), 256, 0, stream>>>(m1, deg, f4acc, m2cnt, m2deg);
    k3_feats<<<512, 256, 0, stream>>>(A, deg, f4acc, m2cnt, m2deg, feat);
    k4_f3<<<512, 256, 0, stream>>>(A, deg, feat);
    k5_stats<<<GG * 7, 256, 0, stream>>>(feat, out);
}
